// BolT_4587025072231
// MI455X (gfx1250) — matrix operand order repaired and checked by reading, not compiled, not run on hardware
//
#include <hip/hip_runtime.h>
#include <hip/hip_bf16.h>
#include <stddef.h>

typedef __attribute__((ext_vector_type(16))) _Float16 v16h;
typedef __attribute__((ext_vector_type(8)))  _Float16 v8h;
typedef __attribute__((ext_vector_type(8)))  float    v8f;

// ---------------------------------------------------------------------------
// helpers
// ---------------------------------------------------------------------------
__device__ __forceinline__ _Float16 to_h(float x)     { return (_Float16)x; }
__device__ __forceinline__ _Float16 to_h(_Float16 x)  { return x; }

__device__ __forceinline__ void cvt_store(float* p, float v)    { *p = v; }
__device__ __forceinline__ void cvt_store(_Float16* p, float v) { *p = (_Float16)v; }

// 16 staged elements as two 16B vectors (f16)
struct H16 { v8h lo, hi; };

__device__ __forceinline__ H16 fetch16(const float* __restrict__ A, int row, int M,
                                       int kbase, int K) {
    H16 r;
    if (row < M && (kbase + 16) <= K) {
        const float4* p = (const float4*)(A + (size_t)row * K + kbase);
        float4 f0 = p[0], f1 = p[1], f2 = p[2], f3 = p[3];
        r.lo = (v8h){ (_Float16)f0.x, (_Float16)f0.y, (_Float16)f0.z, (_Float16)f0.w,
                      (_Float16)f1.x, (_Float16)f1.y, (_Float16)f1.z, (_Float16)f1.w };
        r.hi = (v8h){ (_Float16)f2.x, (_Float16)f2.y, (_Float16)f2.z, (_Float16)f2.w,
                      (_Float16)f3.x, (_Float16)f3.y, (_Float16)f3.z, (_Float16)f3.w };
    } else {
#pragma unroll
        for (int e = 0; e < 16; ++e) {
            const int kk = kbase + e;
            const _Float16 v = (row < M && kk < K)
                ? (_Float16)A[(size_t)row * K + kk] : (_Float16)0.f;
            if (e < 8) r.lo[e] = v; else r.hi[e - 8] = v;
        }
    }
    return r;
}

__device__ __forceinline__ H16 fetch16(const _Float16* __restrict__ A, int row, int M,
                                       int kbase, int K) {
    H16 r;
    if (row < M && (kbase + 16) <= K) {
        const v8h* p = (const v8h*)(A + (size_t)row * K + kbase);
        r.lo = p[0];
        r.hi = p[1];
    } else {
#pragma unroll
        for (int e = 0; e < 16; ++e) {
            const int kk = kbase + e;
            const _Float16 v = (row < M && kk < K)
                ? A[(size_t)row * K + kk] : (_Float16)0.f;
            if (e < 8) r.lo[e] = v; else r.hi[e - 8] = v;
        }
    }
    return r;
}

__device__ __forceinline__ void store16(_Float16* dst /*32B aligned LDS*/, H16 v) {
    ((v8h*)dst)[0] = v.lo;
    ((v8h*)dst)[1] = v.hi;
}

__device__ __forceinline__ v16h combine8(v8h lo, v8h hi) {
    return __builtin_shufflevector(lo, hi, 0,1,2,3,4,5,6,7,8,9,10,11,12,13,14,15);
}

__device__ __forceinline__ v8f wmma_f16(v16h a, v16h b, v8f c) {
    return __builtin_amdgcn_wmma_f32_16x16x32_f16(false, a, false, b, (short)0, c,
                                                  false, false);
}

// ---------------------------------------------------------------------------
// Generic GEMM:  out[M,N] = A[M,K] * W[N,K]^T + bias[N]
// Block tile 128x128, K-step 32, 8 waves (wave32), wave tile 32x64.
// Double-buffered LDS with register prefetch: global loads for tile kt+1
// overlap the 8 WMMAs of tile kt; one barrier per K-step.
// W row-major [N,K] is exactly B^T, so both A and B WMMA fragments are
// contiguous 16B/32B vector loads from row-major LDS tiles (layout-exact
// for the wave32 16x16x32 f16 VGPR striping).
// ---------------------------------------------------------------------------
#define BM 128
#define BN 128
#define BK 32

template <typename AT, typename OT>
__global__ __launch_bounds__(256) void gemm_bias_wmma(
    const AT* __restrict__ A, const float* __restrict__ W,
    const float* __restrict__ bias, OT* __restrict__ out,
    int M, int N, int K)
{
    __shared__ __align__(64) _Float16 sA[2][BM][BK];
    __shared__ __align__(64) _Float16 sB[2][BN][BK];

    const int tid   = threadIdx.x;
    const int lane  = tid & 31;
    const int wv    = tid >> 5;
    const int g     = lane >> 4;       // half-wave group
    const int l15   = lane & 15;
    const int waveM = wv >> 1;         // 0..3  (rows, *32)
    const int waveN = wv & 1;          // 0..1  (cols, *64)
    const int m0    = blockIdx.x * BM;
    const int n0    = blockIdx.y * BN;

    const int rLd = tid >> 1;          // 0..127 : tile row this thread stages
    const int cLd = (tid & 1) * 16;    // 0 or 16: k-offset this thread stages
    const int rowA = m0 + rLd;
    const int rowW = n0 + rLd;

    v8f acc[2][4];
    const v8f vzero = {0.f,0.f,0.f,0.f,0.f,0.f,0.f,0.f};
#pragma unroll
    for (int i = 0; i < 2; ++i)
#pragma unroll
        for (int j = 0; j < 4; ++j) acc[i][j] = vzero;

    const int nK = (K + BK - 1) / BK;

    // prologue: stage tile 0
    store16(&sA[0][rLd][cLd], fetch16(A, rowA, M, cLd, K));
    store16(&sB[0][rLd][cLd], fetch16(W, rowW, N, cLd, K));
    __syncthreads();

    for (int kt = 0; kt < nK; ++kt) {
        const int cur = kt & 1;
        const bool have_next = (kt + 1) < nK;

        // warm GL2 for tile kt+2 (global_prefetch_b8; no counter, no VGPR ret)
        if ((kt + 2) < nK) {
            if (rowA < M) __builtin_prefetch(A + (size_t)rowA * K + (kt + 2) * BK + cLd, 0, 1);
            if (rowW < N) __builtin_prefetch(W + (size_t)rowW * K + (kt + 2) * BK + cLd, 0, 1);
        }

        // register prefetch of tile kt+1 (overlaps WMMA compute below)
        H16 pa, pw;
        if (have_next) {
            const int k0n = (kt + 1) * BK;
            pa = fetch16(A, rowA, M, k0n + cLd, K);
            pw = fetch16(W, rowW, N, k0n + cLd, K);
        }

        // ---- WMMA fragments from LDS (layout-exact contiguous loads) ----
        v16h af[2], bf[4];
#pragma unroll
        for (int i = 0; i < 2; ++i) {
            const int r = waveM * 32 + i * 16 + l15;
            af[i] = combine8(*(const v8h*)&sA[cur][r][g * 8],
                             *(const v8h*)&sA[cur][r][16 + g * 8]);
        }
#pragma unroll
        for (int j = 0; j < 4; ++j) {
            const int r = waveN * 64 + j * 16 + l15;
            bf[j] = combine8(*(const v8h*)&sB[cur][r][g * 8],
                             *(const v8h*)&sB[cur][r][16 + g * 8]);
        }
#pragma unroll
        for (int i = 0; i < 2; ++i)
#pragma unroll
            for (int j = 0; j < 4; ++j)
                acc[i][j] = wmma_f16(af[i], bf[j], acc[i][j]);

        // commit prefetched tile into the alternate buffer
        if (have_next) {
            store16(&sA[cur ^ 1][rLd][cLd], pa);
            store16(&sB[cur ^ 1][rLd][cLd], pw);
        }
        __syncthreads();
    }

    // ---- store D + bias ----
#pragma unroll
    for (int i = 0; i < 2; ++i) {
#pragma unroll
        for (int j = 0; j < 4; ++j) {
            const int mbase = m0 + waveM * 32 + i * 16 + g * 8;
            const int n     = n0 + waveN * 64 + j * 16 + l15;
            if (n < N) {
                const float bv = bias[n];
#pragma unroll
                for (int r = 0; r < 8; ++r) {
                    const int m = mbase + r;
                    if (m < M)
                        cvt_store(&out[(size_t)m * N + n], acc[i][j][r] + bv);
                }
            }
        }
    }
}

// ---------------------------------------------------------------------------
// Fused attention: one wave per (window b, head h). 4 waves/block.
// scores = (q k^T)*scale + bias, boundary mask, softmax, out = P v.
// Both matmuls via v_wmma_f32_16x16x32_f16 with zero-padded LDS tiles.
// ---------------------------------------------------------------------------
__global__ __launch_bounds__(128) void bolt_attention(
    const _Float16* __restrict__ qb, const _Float16* __restrict__ kvb,
    _Float16* __restrict__ att,
    const int* __restrict__ mask_left, const int* __restrict__ mask_right,
    const int* __restrict__ nW_ptr,
    const float* __restrict__ rel_table, const float* __restrict__ cls_up,
    const float* __restrict__ cls_down, const float* __restrict__ cls_self)
{
    __shared__ __align__(64) _Float16 Qs[4][32][32]; // [n][d], zero-padded
    __shared__ __align__(64) _Float16 Ks[4][48][32]; // [m][d], zero-padded
    __shared__ __align__(64) _Float16 Vt[4][32][64]; // [d][m], transposed V
    __shared__ __align__(64) float    Sx[4][32][48]; // scores
    __shared__ __align__(64) _Float16 Ps[4][32][64]; // softmaxed P, zero-padded

    const int b    = blockIdx.x;
    const int w    = threadIdx.x >> 5;
    const int lane = threadIdx.x & 31;
    const int g    = lane >> 4;
    const int l15  = lane & 15;
    const int h    = blockIdx.y * 4 + w;   // 0..35

    // ---- stage q / k / v (f16) into padded LDS tiles ----
    for (int idx = lane; idx < 32 * 32; idx += 32) {
        const int n = idx >> 5, d = idx & 31;
        Qs[w][n][d] = (n < 21 && d < 20)
            ? qb[(size_t)(b * 21 + n) * 720 + h * 20 + d] : (_Float16)0.f;
    }
    for (int idx = lane; idx < 48 * 32; idx += 32) {
        const int m = idx >> 5, d = idx & 31;
        Ks[w][m][d] = (m < 41 && d < 20)
            ? kvb[(size_t)(b * 41 + m) * 1440 + h * 20 + d] : (_Float16)0.f;
    }
    for (int idx = lane; idx < 32 * 64; idx += 32) {
        const int d = idx >> 6, m = idx & 63;
        Vt[w][d][m] = (d < 20 && m < 41)
            ? kvb[(size_t)(b * 41 + m) * 1440 + 720 + h * 20 + d] : (_Float16)0.f;
    }
    __syncthreads();

    // ---- S = Q K^T : 2 (row tiles) x 3 (col tiles) WMMAs ----
    const v8f vzero = {0.f,0.f,0.f,0.f,0.f,0.f,0.f,0.f};
    v16h qa[2];
#pragma unroll
    for (int i = 0; i < 2; ++i) {
        const int r = i * 16 + l15;
        qa[i] = combine8(*(const v8h*)&Qs[w][r][g * 8],
                         *(const v8h*)&Qs[w][r][16 + g * 8]);
    }
    v8f s[2][3];
#pragma unroll
    for (int i = 0; i < 2; ++i)
#pragma unroll
        for (int j = 0; j < 3; ++j) s[i][j] = vzero;
#pragma unroll
    for (int j = 0; j < 3; ++j) {
        const v16h kb = combine8(*(const v8h*)&Ks[w][j * 16 + l15][g * 8],
                                 *(const v8h*)&Ks[w][j * 16 + l15][16 + g * 8]);
#pragma unroll
        for (int i = 0; i < 2; ++i) s[i][j] = wmma_f16(qa[i], kb, s[i][j]);
    }

    // ---- scale + relative/cls bias + boundary masks, spill scores to LDS ----
    const int nWv = nW_ptr[0];
    const int win = b % nWv;
    const int mc  = nWv < 2 ? nWv : 2;
    const int* mlp = (win < mc) ? (mask_left + win * 861) : (const int*)0;
    const int* mrp = (win >= nWv - mc)
        ? (mask_right + ((2 - mc) + (win - (nWv - mc))) * 861) : (const int*)0;
    const float scale = 0.22360679774997896f; // 20^-0.5
#pragma unroll
    for (int i = 0; i < 2; ++i) {
#pragma unroll
        for (int j = 0; j < 3; ++j) {
#pragma unroll
            for (int r = 0; r < 8; ++r) {
                const int qn = i * 16 + g * 8 + r;
                const int am = j * 16 + l15;
                float val = 0.f;
                if (qn < 21 && am < 41) {
                    float bv;
                    if (qn == 0)
                        bv = (am == 0) ? cls_self[h] : cls_up[h * 40 + am - 1];
                    else
                        bv = (am == 0) ? cls_down[h * 20 + qn - 1]
                                       : rel_table[((qn - 1) - (am - 1) + 39) * 36 + h];
                    val = s[i][j][r] * scale + bv;
                    if (mlp && mlp[qn * 41 + am]) val = -3.402823466e38f;
                    if (mrp && mrp[qn * 41 + am]) val = -3.402823466e38f;
                }
                Sx[w][qn][am] = val;
            }
        }
    }
    __syncthreads();

    // ---- per-row softmax (row == lane), build zero-padded P (f16) ----
    if (lane < 21) {
        float mx = -3.402823466e38f;
        for (int am = 0; am < 41; ++am) mx = fmaxf(mx, Sx[w][lane][am]);
        float sum = 0.f;
        for (int am = 0; am < 41; ++am) {
            const float e = __expf(Sx[w][lane][am] - mx);
            sum += e;
            Sx[w][lane][am] = e;
        }
        const float inv = 1.0f / sum;
        for (int am = 0; am < 41; ++am)
            Ps[w][lane][am] = (_Float16)(Sx[w][lane][am] * inv);
        for (int am = 41; am < 64; ++am) Ps[w][lane][am] = (_Float16)0.f;
    } else {
        for (int am = 0; am < 64; ++am) Ps[w][lane][am] = (_Float16)0.f;
    }
    __syncthreads();

    // ---- O = P V : 2 row tiles x 2 d tiles x 2 K chunks = 8 WMMAs ----
    v8f o[2][2];
#pragma unroll
    for (int i = 0; i < 2; ++i)
#pragma unroll
        for (int j = 0; j < 2; ++j) o[i][j] = vzero;
#pragma unroll
    for (int kc = 0; kc < 2; ++kc) {
        v16h pa[2];
#pragma unroll
        for (int i = 0; i < 2; ++i) {
            const int r = i * 16 + l15;
            pa[i] = combine8(*(const v8h*)&Ps[w][r][kc * 32 + g * 8],
                             *(const v8h*)&Ps[w][r][kc * 32 + 16 + g * 8]);
        }
#pragma unroll
        for (int j = 0; j < 2; ++j) {
            const v16h vb = combine8(*(const v8h*)&Vt[w][j * 16 + l15][kc * 32 + g * 8],
                                     *(const v8h*)&Vt[w][j * 16 + l15][kc * 32 + 16 + g * 8]);
#pragma unroll
            for (int i = 0; i < 2; ++i) o[i][j] = wmma_f16(pa[i], vb, o[i][j]);
        }
    }

    // ---- store head output (f16) into [B_*21, 720] ----
#pragma unroll
    for (int i = 0; i < 2; ++i) {
#pragma unroll
        for (int j = 0; j < 2; ++j) {
#pragma unroll
            for (int r = 0; r < 8; ++r) {
                const int qn = i * 16 + g * 8 + r;
                const int d  = j * 16 + l15;
                if (qn < 21 && d < 20)
                    att[(size_t)(b * 21 + qn) * 720 + h * 20 + d] =
                        (_Float16)o[i][j][r];
            }
        }
    }
}

// ---------------------------------------------------------------------------
// launcher
// ---------------------------------------------------------------------------
extern "C" void kernel_launch(void* const* d_in, const int* in_sizes, int n_in,
                              void* d_out, int out_size, void* d_ws, size_t ws_size,
                              hipStream_t stream) {
    const float* x          = (const float*)d_in[0];
    const float* x_         = (const float*)d_in[1];
    const int*   mask_left  = (const int*)d_in[2];
    const int*   mask_right = (const int*)d_in[3];
    const int*   nW         = (const int*)d_in[4];
    const float* Wq         = (const float*)d_in[5];
    const float* bq         = (const float*)d_in[6];
    const float* Wkv        = (const float*)d_in[7];
    const float* bkv        = (const float*)d_in[8];
    const float* Wproj      = (const float*)d_in[9];
    const float* bproj      = (const float*)d_in[10];
    const float* rel_table  = (const float*)d_in[11];
    const float* cls_up     = (const float*)d_in[12];
    const float* cls_down   = (const float*)d_in[13];
    const float* cls_self   = (const float*)d_in[14];

    const int C = 400, HD = 720, KV2 = 1440;
    const int B_ = in_sizes[0] / (21 * C);   // 3200
    const int M1 = B_ * 21;                  // q / output rows
    const int M2 = B_ * 41;                  // kv rows

    auto align_up = [](size_t v) { return (v + 255) & ~(size_t)255; };
    const size_t qbytes  = align_up((size_t)M1 * HD  * sizeof(_Float16));
    const size_t kvbytes = align_up((size_t)M2 * KV2 * sizeof(_Float16));
    const size_t atbytes = align_up((size_t)M1 * HD  * sizeof(_Float16));
    if (ws_size < qbytes + kvbytes + atbytes) return;

    char* ws = (char*)d_ws;
    _Float16* qbuf  = (_Float16*)ws;
    _Float16* kvbuf = (_Float16*)(ws + qbytes);
    _Float16* atbuf = (_Float16*)(ws + qbytes + kvbytes);

    dim3 blk(256);
    // q = x @ Wq^T + bq          (f32 in, f16 out)
    gemm_bias_wmma<float, _Float16>
        <<<dim3((M1 + BM - 1) / BM, (HD + BN - 1) / BN), blk, 0, stream>>>(
            x, Wq, bq, qbuf, M1, HD, C);
    // kv = x_ @ Wkv^T + bkv
    gemm_bias_wmma<float, _Float16>
        <<<dim3((M2 + BM - 1) / BM, (KV2 + BN - 1) / BN), blk, 0, stream>>>(
            x_, Wkv, bkv, kvbuf, M2, KV2, C);
    // fused attention: grid (window, head-group of 4), 36 heads -> y = 9
    bolt_attention<<<dim3(B_, 9), 128, 0, stream>>>(
        qbuf, kvbuf, atbuf, mask_left, mask_right, nW,
        rel_table, cls_up, cls_down, cls_self);
    // out = att @ Wproj^T + bproj (f16 in, f32 out)
    gemm_bias_wmma<_Float16, float>
        <<<dim3((M1 + BM - 1) / BM, (C + BN - 1) / BN), blk, 0, stream>>>(
            atbuf, Wproj, bproj, (float*)d_out, M1, C, HD);
}
